// DAGNN_85770496901352
// MI455X (gfx1250) — compile-verified, matrix operand order checked by reading, not run on hardware
//
#include <hip/hip_runtime.h>
#include <hip/hip_bf16.h>

// ---------------------------------------------------------------------------
// Types for CDNA5 WMMA (wave32): A/B = 16 x f16 per lane, C/D = 8 x f32.
// ---------------------------------------------------------------------------
typedef __attribute__((ext_vector_type(16))) _Float16 v16h;
typedef __attribute__((ext_vector_type(8)))  _Float16 v8h;
typedef __attribute__((ext_vector_type(8)))  float    v8f;

#define N_NODES  100000
#define T_STEPS  8
#define N_EDGES  1600000
#define E_TOT    (N_EDGES + N_NODES)   // with self loops
#define N_GRAPHS 128

#define BSTRIDE 40   // halfs per LDS column: 32 data + 8 pad (80B, 16B aligned)

// ---------------------------------------------------------------------------
// Generic WMMA GEMM: C[M,N] = A[M,K] * B + bias, optional relu.
// A row-major (lda).
//   KCONTIG=true : B element (k,n) at B[n*ldb + k]  (transposed torch weight)
//   KCONTIG=false: B element (k,n) at B[k*ldb + n]  (row-major weight)
// blockDim = 64 (2 waves); wave computes a 16(M) x 64(N) tile.
// grid = (M/32, N/64). K % 32 == 0, M % 32 == 0, N % 64 == 0 (true here).
// ---------------------------------------------------------------------------
template <bool KCONTIG>
__global__ void dagnn_gemm_wmma_f16(const float* __restrict__ A,
                                    const float* __restrict__ B,
                                    const float* __restrict__ bias,
                                    float* __restrict__ C,
                                    int M, int N, int K,
                                    int lda, int ldb, int ldc,
                                    int relu) {
  __shared__ __align__(16) _Float16 smemB[64 * BSTRIDE];

  const int tid  = threadIdx.x;
  const int wave = tid >> 5;
  const int lane = tid & 31;
  const int laneHalf = lane >> 4;          // 0 or 1
  const int lm = lane & 15;
  const int m0 = (blockIdx.x * 2 + wave) * 16;
  const int n0 = blockIdx.y * 64;

  v8f acc0 = {}, acc1 = {}, acc2 = {}, acc3 = {};

  // lane's A row base: row m0+lm, starting at k offset 8*laneHalf
  const float* arow = A + (size_t)(m0 + lm) * lda + laneHalf * 8;

  for (int k0 = 0; k0 < K; k0 += 32) {
    // ---------------- stage B tile (32k x 64n) into LDS as f16 [n][k] ------
    if (KCONTIG) {
      // thread t owns column n0+t; k contiguous in memory: 8x b128 loads
      const float* bp = B + (size_t)(n0 + tid) * ldb + k0;
      _Float16 hcol[32];
#pragma unroll
      for (int q = 0; q < 8; ++q) {
        const float4 f = *(const float4*)(bp + q * 4);
        hcol[q * 4 + 0] = (_Float16)f.x;
        hcol[q * 4 + 1] = (_Float16)f.y;
        hcol[q * 4 + 2] = (_Float16)f.z;
        hcol[q * 4 + 3] = (_Float16)f.w;
      }
      v8h* dst = (v8h*)(smemB + tid * BSTRIDE);
#pragma unroll
      for (int q = 0; q < 4; ++q) {
        v8h v;
#pragma unroll
        for (int j = 0; j < 8; ++j) v[j] = hcol[q * 8 + j];
        dst[q] = v;                       // ds_store_b128
      }
    } else {
      // row-major B: fetch tile as 512 float4 chunks (8 per thread, coalesced)
      // chunk c covers (kk = c>>4, nn = (c&15)*4 .. +3)
#pragma unroll
      for (int i = 0; i < 8; ++i) {
        const int c  = i * 64 + tid;
        const int kk = c >> 4;
        const int nn = (c & 15) * 4;
        const float4 f = *(const float4*)(B + (size_t)(k0 + kk) * ldb + n0 + nn);
        smemB[(nn + 0) * BSTRIDE + kk] = (_Float16)f.x;   // ds_store_b16
        smemB[(nn + 1) * BSTRIDE + kk] = (_Float16)f.y;
        smemB[(nn + 2) * BSTRIDE + kk] = (_Float16)f.z;
        smemB[(nn + 3) * BSTRIDE + kk] = (_Float16)f.w;
      }
    }

    // ---------------- A fragment: 4x b128 global loads ---------------------
    // k(j) = k0 + 8*laneHalf + (j&7) + 16*(j>>3)
    const float* ar = arow + k0;
    const float4 a0 = *(const float4*)(ar + 0);
    const float4 a1 = *(const float4*)(ar + 4);
    const float4 a2 = *(const float4*)(ar + 16);
    const float4 a3 = *(const float4*)(ar + 20);
    v16h a;
    a[0]  = (_Float16)a0.x; a[1]  = (_Float16)a0.y;
    a[2]  = (_Float16)a0.z; a[3]  = (_Float16)a0.w;
    a[4]  = (_Float16)a1.x; a[5]  = (_Float16)a1.y;
    a[6]  = (_Float16)a1.z; a[7]  = (_Float16)a1.w;
    a[8]  = (_Float16)a2.x; a[9]  = (_Float16)a2.y;
    a[10] = (_Float16)a2.z; a[11] = (_Float16)a2.w;
    a[12] = (_Float16)a3.x; a[13] = (_Float16)a3.y;
    a[14] = (_Float16)a3.z; a[15] = (_Float16)a3.w;

    __syncthreads();   // B tile visible

    // ---------------- 4 WMMAs, B frags from LDS (2x ds_load_b128 each) -----
#pragma unroll
    for (int t = 0; t < 4; ++t) {
      const _Float16* bp = smemB + (t * 16 + lm) * BSTRIDE + laneHalf * 8;
      const v8h lo = *(const v8h*)(bp);        // k = 8*laneHalf + 0..7
      const v8h hi = *(const v8h*)(bp + 16);   // k = 16 + 8*laneHalf + 0..7
      v16h b;
#pragma unroll
      for (int j = 0; j < 8; ++j) { b[j] = lo[j]; b[8 + j] = hi[j]; }
      v8f c = (t == 0) ? acc0 : (t == 1) ? acc1 : (t == 2) ? acc2 : acc3;
      c = __builtin_amdgcn_wmma_f32_16x16x32_f16(false, a, false, b,
                                                 (short)0, c, false, false);
      if (t == 0) acc0 = c; else if (t == 1) acc1 = c;
      else if (t == 2) acc2 = c; else acc3 = c;
    }

    __syncthreads();   // done reading B tile before next overwrite
  }

  // ---- epilogue: VGPR r holds (m = m0 + r + 8*laneHalf, n = lane%16)
#pragma unroll
  for (int t = 0; t < 4; ++t) {
    const v8f c = (t == 0) ? acc0 : (t == 1) ? acc1 : (t == 2) ? acc2 : acc3;
    const int n = n0 + t * 16 + lm;
    const float bv = bias ? bias[n] : 0.0f;
    float* cp = C + (size_t)(m0 + laneHalf * 8) * ldc + n;
#pragma unroll
    for (int r = 0; r < 8; ++r) {
      float v = c[r] + bv;
      if (relu) v = fmaxf(v, 0.0f);
      cp[(size_t)r * ldc] = v;
    }
  }
}

// ---------------------------------------------------------------------------
// Fill kernels (graph-capture-safe init)
// ---------------------------------------------------------------------------
__global__ void dagnn_fill_f32(float* p, float v, int n) {
  int i = blockIdx.x * blockDim.x + threadIdx.x;
  if (i < n) p[i] = v;
}
__global__ void dagnn_fill_u32(unsigned* p, unsigned v, int n) {
  int i = blockIdx.x * blockDim.x + threadIdx.x;
  if (i < n) p[i] = v;
}

// ---------------------------------------------------------------------------
// GRU gate update for one timestep: h = (1-z)*n + z*h
// ---------------------------------------------------------------------------
__global__ void dagnn_gru_gate(const float* __restrict__ gi,
                               const float* __restrict__ gh,
                               float* __restrict__ h, int t) {
  int idx = blockIdx.x * blockDim.x + threadIdx.x;
  if (idx >= N_NODES * 64) return;
  int n = idx >> 6, j = idx & 63;
  const float* g    = gi + ((size_t)n * T_STEPS + t) * 192;
  const float* gh_n = gh + (size_t)n * 192;
  float r = 1.0f / (1.0f + __expf(-(g[j]      + gh_n[j])));
  float z = 1.0f / (1.0f + __expf(-(g[64 + j] + gh_n[64 + j])));
  float nn = tanhf(g[128 + j] + r * gh_n[128 + j]);
  float hv = h[(size_t)n * 64 + j];
  h[(size_t)n * 64 + j] = (1.0f - z) * nn + z * hv;
}

// ---------------------------------------------------------------------------
// GAT helpers
// ---------------------------------------------------------------------------
__device__ __forceinline__ void edge_sd(const int* __restrict__ ei, int e,
                                        int& s, int& d) {
  if (e < N_EDGES) { s = ei[e]; d = ei[N_EDGES + e]; }
  else             { s = d = e - N_EDGES; }
}
__device__ __forceinline__ unsigned f2ord(float f) {
  unsigned u = __float_as_uint(f);
  return (u & 0x80000000u) ? ~u : (u | 0x80000000u);
}
__device__ __forceinline__ float ord2f(unsigned u) {
  return (u & 0x80000000u) ? __uint_as_float(u & 0x7fffffffu)
                           : __uint_as_float(~u);
}

__global__ void dagnn_gat_alpha(const float* __restrict__ xs,
                                const float* __restrict__ att_src,
                                const float* __restrict__ att_dst,
                                float* __restrict__ a_src,
                                float* __restrict__ a_dst, int H) {
  int idx = blockIdx.x * blockDim.x + threadIdx.x;   // n*H + h
  if (idx >= N_NODES * H) return;
  int hh = idx % H;
  const float* xrow = xs + (size_t)idx * 64;
  float s = 0.f, d = 0.f;
#pragma unroll 8
  for (int c = 0; c < 64; ++c) {
    float v = xrow[c];
    s += v * att_src[hh * 64 + c];
    d += v * att_dst[hh * 64 + c];
  }
  a_src[idx] = s; a_dst[idx] = d;
}

__global__ void dagnn_gat_logits(const int* __restrict__ ei,
                                 const float* __restrict__ a_src,
                                 const float* __restrict__ a_dst,
                                 float* __restrict__ e_out,
                                 unsigned* __restrict__ emax, int H) {
  int idx = blockIdx.x * blockDim.x + threadIdx.x;   // e*H + h
  if (idx >= E_TOT * H) return;
  int e = idx / H, hh = idx % H;
  int s, d; edge_sd(ei, e, s, d);
  float v = a_src[s * H + hh] + a_dst[d * H + hh];
  v = (v > 0.f) ? v : 0.2f * v;
  e_out[idx] = v;
  atomicMax(&emax[d * H + hh], f2ord(v));
}

__global__ void dagnn_gat_expsum(const int* __restrict__ ei,
                                 const float* __restrict__ e_in,
                                 const unsigned* __restrict__ emax,
                                 float* __restrict__ ee,
                                 float* __restrict__ denom, int H) {
  int idx = blockIdx.x * blockDim.x + threadIdx.x;
  if (idx >= E_TOT * H) return;
  int e = idx / H, hh = idx % H;
  int s, d; edge_sd(ei, e, s, d);
  float x = __expf(e_in[idx] - ord2f(emax[d * H + hh]));
  ee[idx] = x;
  atomicAdd(&denom[d * H + hh], x);
}

__global__ void dagnn_gat_msg(const int* __restrict__ ei,
                              const float* __restrict__ xs,
                              const float* __restrict__ ee,
                              const float* __restrict__ denom,
                              float* __restrict__ out, int H) {
  int idx = blockIdx.x * blockDim.x + threadIdx.x;
  if (idx >= E_TOT * H) return;
  int e = idx / H, hh = idx % H;
  int s, d; edge_sd(ei, e, s, d);
  float alpha = ee[idx] / (denom[d * H + hh] + 1e-16f);
  const float* xrow = xs + ((size_t)s * H + hh) * 64;
  float* orow = out + ((size_t)d * H + hh) * 64;
#pragma unroll 4
  for (int c = 0; c < 64; ++c) atomicAdd(&orow[c], alpha * xrow[c]);
}

__global__ void dagnn_bias_relu(float* __restrict__ out,
                                const float* __restrict__ bias, int n, int D) {
  int i = blockIdx.x * blockDim.x + threadIdx.x;
  if (i < n) out[i] = fmaxf(out[i] + bias[i % D], 0.0f);
}

// ---------------------------------------------------------------------------
// mean pool per graph (attn softmax over axis=1 of [N,1] is identically 1.0)
// ---------------------------------------------------------------------------
__global__ void dagnn_pool(const float* __restrict__ h,
                           const int* __restrict__ batch,
                           float* __restrict__ gsum, float* __restrict__ gcnt) {
  int n = blockIdx.x * blockDim.x + threadIdx.x;
  if (n >= N_NODES) return;
  int g = batch[n];
#pragma unroll 4
  for (int c = 0; c < 64; ++c)
    atomicAdd(&gsum[g * 64 + c], h[(size_t)n * 64 + c]);
  atomicAdd(&gcnt[g], 1.0f);
}

__global__ void dagnn_head(const float* __restrict__ gsum,
                           const float* __restrict__ gcnt,
                           const float* __restrict__ fc_w,
                           const float* __restrict__ fc_b,
                           float* __restrict__ out) {
  int g = blockIdx.x * blockDim.x + threadIdx.x;
  if (g >= N_GRAPHS) return;
  float inv = 1.0f / fmaxf(gcnt[g], 1.0f);
  float logits[10];
#pragma unroll
  for (int o = 0; o < 10; ++o) {
    float acc = fc_b[o];
    for (int c = 0; c < 64; ++c)
      acc += (gsum[g * 64 + c] * inv) * fc_w[c * 10 + o];
    logits[o] = acc;
  }
  float m = logits[0];
#pragma unroll
  for (int o = 1; o < 10; ++o) m = fmaxf(m, logits[o]);
  float s = 0.f;
#pragma unroll
  for (int o = 0; o < 10; ++o) s += __expf(logits[o] - m);
  float ls = logf(s);
#pragma unroll
  for (int o = 0; o < 10; ++o) out[g * 10 + o] = logits[o] - m - ls;
}

// ---------------------------------------------------------------------------
// Host orchestration.  Workspace offsets in floats.
// Peak: gi + h + gh during GRU = 179.2M floats (~717 MB).
// ---------------------------------------------------------------------------
#define OFF_GI    ((size_t)0)             // [800000,192]      153.6M
#define OFF_H     ((size_t)153600000)     // [100000,64]         6.4M (persists)
#define OFF_GH    ((size_t)160000000)     // [100000,192]       19.2M
// Post-GRU, reuse the gi region:
#define OFF_XS1   ((size_t)0)             // [N,4,64]           25.6M
#define OFF_H2    ((size_t)25600000)      // [N,256]            25.6M
#define OFF_E1    ((size_t)51200000)      // [Etot,4]            6.8M
#define OFF_EE1   ((size_t)58000000)      // [Etot,4]            6.8M
#define OFF_AS1   ((size_t)64800000)      // [N,4]
#define OFF_AD1   ((size_t)65200000)
#define OFF_EMX1  ((size_t)65600000)      // uint
#define OFF_DEN1  ((size_t)66000000)
#define OFF_XS2   ((size_t)66400000)      // [N,64]
#define OFF_H3    ((size_t)72800000)      // [N,64]
#define OFF_AS2   ((size_t)79200000)
#define OFF_AD2   ((size_t)79300000)
#define OFF_EMX2  ((size_t)79400000)
#define OFF_DEN2  ((size_t)79500000)
#define OFF_E2    ((size_t)79600000)      // [Etot,1]
#define OFF_EE2   ((size_t)81300000)
#define OFF_GSUM  ((size_t)83000000)      // [128,64]
#define OFF_GCNT  ((size_t)83008192)      // [128]

extern "C" void kernel_launch(void* const* d_in, const int* in_sizes, int n_in,
                              void* d_out, int out_size, void* d_ws, size_t ws_size,
                              hipStream_t stream) {
  const float* x     = (const float*)d_in[0];
  const int*   ei    = (const int*)d_in[1];
  const int*   batch = (const int*)d_in[2];
  const float* w_ih  = (const float*)d_in[4];   // [192,128]
  const float* w_hh  = (const float*)d_in[5];   // [192,64]
  const float* b_ih  = (const float*)d_in[6];
  const float* b_hh  = (const float*)d_in[7];
  const float* W1    = (const float*)d_in[8];   // [64,256]
  const float* as1   = (const float*)d_in[9];
  const float* ad1   = (const float*)d_in[10];
  const float* b1    = (const float*)d_in[11];
  const float* W2    = (const float*)d_in[12];  // [256,64]
  const float* as2   = (const float*)d_in[13];
  const float* ad2   = (const float*)d_in[14];
  const float* b2    = (const float*)d_in[15];
  const float* fc_w  = (const float*)d_in[18];
  const float* fc_b  = (const float*)d_in[19];
  float* ws  = (float*)d_ws;
  float* out = (float*)d_out;

  // KCONTIG=true: B(k,n) = B[n*ldb + k]   (torch [out,in] weight used as W^T)
  auto gemmT = [&](const float* A, const float* B, const float* bias, float* C,
                   int M, int N, int K, int lda, int ldb, int ldc, int relu) {
    dim3 grid(M / 32, N / 64);
    dagnn_gemm_wmma_f16<true><<<grid, dim3(64), 0, stream>>>(
        A, B, bias, C, M, N, K, lda, ldb, ldc, relu);
  };
  // KCONTIG=false: B(k,n) = B[k*ldb + n]  (row-major [K,N] weight)
  auto gemmN = [&](const float* A, const float* B, const float* bias, float* C,
                   int M, int N, int K, int lda, int ldb, int ldc, int relu) {
    dim3 grid(M / 32, N / 64);
    dagnn_gemm_wmma_f16<false><<<grid, dim3(64), 0, stream>>>(
        A, B, bias, C, M, N, K, lda, ldb, ldc, relu);
  };
  auto fillf = [&](float* p, float v, int n) {
    dagnn_fill_f32<<<(n + 255) / 256, 256, 0, stream>>>(p, v, n);
  };
  auto fillu = [&](unsigned* p, unsigned v, int n) {
    dagnn_fill_u32<<<(n + 255) / 256, 256, 0, stream>>>(p, v, n);
  };

  // ---- GRU -----------------------------------------------------------------
  // gi = x @ w_ih^T + b_ih over [N*T,128]x[128,192]
  gemmT(x, w_ih, b_ih, ws + OFF_GI, N_NODES * T_STEPS, 192, 128,
        128, /*ldb*/128, 192, 0);
  fillf(ws + OFF_H, 0.0f, N_NODES * 64);
  for (int t = 0; t < T_STEPS; ++t) {
    // gh = h @ w_hh^T + b_hh
    gemmT(ws + OFF_H, w_hh, b_hh, ws + OFF_GH, N_NODES, 192, 64,
          64, /*ldb*/64, 192, 0);
    int nt = N_NODES * 64;
    dagnn_gru_gate<<<(nt + 255) / 256, 256, 0, stream>>>(
        ws + OFF_GI, ws + OFF_GH, ws + OFF_H, t);
  }

  // ---- GAT layer 1 (H=4, C=64, concat -> 256) ------------------------------
  gemmN(ws + OFF_H, W1, nullptr, ws + OFF_XS1, N_NODES, 256, 64,
        64, /*ldb*/256, 256, 0);
  {
    int H = 4;
    int nh = N_NODES * H;
    dagnn_gat_alpha<<<(nh + 255) / 256, 256, 0, stream>>>(
        ws + OFF_XS1, as1, ad1, ws + OFF_AS1, ws + OFF_AD1, H);
    fillu((unsigned*)(ws + OFF_EMX1), 0u, nh);
    fillf(ws + OFF_DEN1, 0.0f, nh);
    fillf(ws + OFF_H2, 0.0f, N_NODES * 256);
    int eh = E_TOT * H;
    dagnn_gat_logits<<<(eh + 255) / 256, 256, 0, stream>>>(
        ei, ws + OFF_AS1, ws + OFF_AD1, ws + OFF_E1,
        (unsigned*)(ws + OFF_EMX1), H);
    dagnn_gat_expsum<<<(eh + 255) / 256, 256, 0, stream>>>(
        ei, ws + OFF_E1, (unsigned*)(ws + OFF_EMX1), ws + OFF_EE1,
        ws + OFF_DEN1, H);
    dagnn_gat_msg<<<(eh + 255) / 256, 256, 0, stream>>>(
        ei, ws + OFF_XS1, ws + OFF_EE1, ws + OFF_DEN1, ws + OFF_H2, H);
    int nd = N_NODES * 256;
    dagnn_bias_relu<<<(nd + 255) / 256, 256, 0, stream>>>(ws + OFF_H2, b1, nd, 256);
  }

  // ---- GAT layer 2 (H=1, C=64, mean -> 64) ---------------------------------
  gemmN(ws + OFF_H2, W2, nullptr, ws + OFF_XS2, N_NODES, 64, 256,
        256, /*ldb*/64, 64, 0);
  {
    int H = 1;
    int nh = N_NODES;
    dagnn_gat_alpha<<<(nh + 255) / 256, 256, 0, stream>>>(
        ws + OFF_XS2, as2, ad2, ws + OFF_AS2, ws + OFF_AD2, H);
    fillu((unsigned*)(ws + OFF_EMX2), 0u, nh);
    fillf(ws + OFF_DEN2, 0.0f, nh);
    fillf(ws + OFF_H3, 0.0f, N_NODES * 64);
    int eh = E_TOT;
    dagnn_gat_logits<<<(eh + 255) / 256, 256, 0, stream>>>(
        ei, ws + OFF_AS2, ws + OFF_AD2, ws + OFF_E2,
        (unsigned*)(ws + OFF_EMX2), H);
    dagnn_gat_expsum<<<(eh + 255) / 256, 256, 0, stream>>>(
        ei, ws + OFF_E2, (unsigned*)(ws + OFF_EMX2), ws + OFF_EE2,
        ws + OFF_DEN2, H);
    dagnn_gat_msg<<<(eh + 255) / 256, 256, 0, stream>>>(
        ei, ws + OFF_XS2, ws + OFF_EE2, ws + OFF_DEN2, ws + OFF_H3, H);
    int nd = N_NODES * 64;
    dagnn_bias_relu<<<(nd + 255) / 256, 256, 0, stream>>>(ws + OFF_H3, b2, nd, 64);
  }

  // ---- pooling + FC + log_softmax -----------------------------------------
  fillf(ws + OFF_GSUM, 0.0f, N_GRAPHS * 64);
  fillf(ws + OFF_GCNT, 0.0f, N_GRAPHS);
  dagnn_pool<<<(N_NODES + 255) / 256, 256, 0, stream>>>(
      ws + OFF_H3, batch, ws + OFF_GSUM, ws + OFF_GCNT);
  dagnn_head<<<1, N_GRAPHS, 0, stream>>>(ws + OFF_GSUM, ws + OFF_GCNT,
                                         fc_w, fc_b, out);
}
